// get_model_49185965474291
// MI455X (gfx1250) — compile-verified, matrix operand order checked by reading, not run on hardware
//
#include <hip/hip_runtime.h>
#include <hip/hip_bf16.h>
#include <math.h>

// ---------------------------------------------------------------------------
// PointNet++-style network for MI455X (gfx1250, wave32).
// MLP einsums run as f16 WMMA tiles (v_wmma_f32_16x16x32_f16) with f32
// accumulation. GEMM blocks are 4 waves computing a 64x16 output panel; the
// 16xK weight panel is staged into LDS once per block with
// global_load_async_to_lds_b128 (ASYNCcnt) and read back as B fragments with
// ds_load_b128, while A streams from global with prefetch.
// Pairwise distances are recomputed from coordinates on the fly (never
// materializing the 8192x8192 / 256MB matrix). FPS runs as a single
// workgroup with dmin resident in LDS.
// ---------------------------------------------------------------------------

typedef _Float16 half_t;
typedef __attribute__((ext_vector_type(16))) _Float16 v16h;
typedef __attribute__((ext_vector_type(8)))  _Float16 v8h;
typedef __attribute__((ext_vector_type(8)))  float    v8f;

#define EPSN 1e-5f

// ----------------------------- small helpers -------------------------------

__global__ void concat_p0_kernel(const float* __restrict__ v,
                                 const float* __restrict__ nrm,
                                 float* __restrict__ p0, int n) {
  int i = blockIdx.x * blockDim.x + threadIdx.x;
  if (i >= n * 6) return;
  int pt = i / 6, c = i % 6;
  p0[i] = (c < 3) ? v[pt * 3 + c] : nrm[pt * 3 + (c - 3)];
}

__global__ void zero_h_kernel(half_t* __restrict__ p, size_t cnt) {
  size_t i = (size_t)blockIdx.x * blockDim.x + threadIdx.x;
  size_t stride = (size_t)gridDim.x * blockDim.x;
  for (; i < cnt; i += stride) p[i] = (half_t)0.f;
}

// weights f32 [Cout x Cin] -> f16 [Cout x Kp], zero padded in K
__global__ void convert_w_kernel(const float* __restrict__ w,
                                 half_t* __restrict__ wh,
                                 int Cout, int Cin, int Kp) {
  int i = blockIdx.x * blockDim.x + threadIdx.x;
  if (i >= Cout * Kp) return;
  int c = i / Kp, k = i % Kp;
  wh[i] = (half_t)((k < Cin) ? w[(size_t)c * Cin + k] : 0.f);
}

// ------------------------------ FPS (1 block) ------------------------------
// Sequential farthest-point sampling; dmin kept in LDS (n <= 8192 -> 32KB).
__global__ __launch_bounds__(1024)
void fps_kernel(const float* __restrict__ xyz, int n, int npoint,
                int* __restrict__ idx_out, float* __restrict__ xyz_out) {
  __shared__ float dmin[8192];
  __shared__ float red_v[1024];
  __shared__ int   red_i[1024];
  __shared__ float cpt[3];
  __shared__ int   s_far;
  int tid = threadIdx.x;
  for (int i = tid; i < n; i += 1024) dmin[i] = 1e10f;
  if (tid == 0) s_far = 0;
  __syncthreads();
  for (int it = 0; it < npoint; ++it) {
    int far = s_far;
    if (tid == 0) {
      idx_out[it] = far;
      cpt[0] = xyz[far * 3 + 0];
      cpt[1] = xyz[far * 3 + 1];
      cpt[2] = xyz[far * 3 + 2];
      xyz_out[it * 3 + 0] = cpt[0];
      xyz_out[it * 3 + 1] = cpt[1];
      xyz_out[it * 3 + 2] = cpt[2];
    }
    __syncthreads();
    float fx = cpt[0], fy = cpt[1], fz = cpt[2];
    float best = -1.f; int bi = 0;
    for (int i = tid; i < n; i += 1024) {
      float dx = xyz[i * 3 + 0] - fx;
      float dy = xyz[i * 3 + 1] - fy;
      float dz = xyz[i * 3 + 2] - fz;
      float d = dx * dx + dy * dy + dz * dz;
      float dm = fminf(dmin[i], d);
      dmin[i] = dm;
      if (dm > best) { best = dm; bi = i; }
    }
    red_v[tid] = best; red_i[tid] = bi;
    __syncthreads();
    for (int o = 512; o > 0; o >>= 1) {
      if (tid < o) {
        if (red_v[tid + o] > red_v[tid] ||
            (red_v[tid + o] == red_v[tid] && red_i[tid + o] < red_i[tid])) {
          red_v[tid] = red_v[tid + o];
          red_i[tid] = red_i[tid + o];
        }
      }
      __syncthreads();
    }
    if (tid == 0) s_far = red_i[0];
    __syncthreads();
  }
}

// ---------------------------- ball query + gather --------------------------
// First 32 qualifying indices in ascending order (matches top_k(-gi) trick).
__global__ void ball_query_kernel(const float* __restrict__ xyz,
                                  const float* __restrict__ new_xyz,
                                  int n, int s, float r2,
                                  int* __restrict__ gidx) {
  int g = blockIdx.x * blockDim.x + threadIdx.x;
  if (g >= s) return;
  float cx = new_xyz[g * 3 + 0], cy = new_xyz[g * 3 + 1], cz = new_xyz[g * 3 + 2];
  int cnt = 0, first = -1;
  for (int j = 0; j < n && cnt < 32; ++j) {
    float dx = xyz[j * 3 + 0] - cx;
    float dy = xyz[j * 3 + 1] - cy;
    float dz = xyz[j * 3 + 2] - cz;
    float d = dx * dx + dy * dy + dz * dz;
    if (d <= r2) {
      if (first < 0) first = j;
      gidx[g * 32 + cnt] = j;
      ++cnt;
    }
  }
  if (first < 0) first = 0;
  for (; cnt < 32; ++cnt) gidx[g * 32 + cnt] = first;
}

// Build f16 GEMM input rows: [grouped_xyz(3) | points[idx](C)] (A pre-zeroed)
__global__ void group_gather_kernel(const float* __restrict__ xyz,
                                    const float* __restrict__ new_xyz,
                                    const float* __restrict__ points,
                                    const int* __restrict__ gidx,
                                    int s, int C, int Kp,
                                    half_t* __restrict__ A) {
  int row = blockIdx.x;              // 0 .. s*32-1
  int g = row >> 5;
  int j = gidx[row];
  half_t* a = A + (size_t)row * Kp;
  for (int c = threadIdx.x; c < C + 3; c += blockDim.x) {
    float v;
    if (c < 3) v = xyz[j * 3 + c] - new_xyz[g * 3 + c];
    else       v = points[(size_t)j * C + (c - 3)];
    a[c] = (half_t)v;
  }
}

// ------------------------------- WMMA GEMM ---------------------------------
// Y[M x Cout] = A[M x K](f16) * W[Cout x K](f16)^T + bias.
// Block = 4 waves -> 64(M) x 16(N) output panel. The 16xK weight panel is
// DMA'd to LDS once (global_load_async_to_lds_b128, ASYNCcnt), then each wave
// reads B fragments from LDS (ds_load_b128) while streaming A from global.
__global__ __launch_bounds__(128)
void gemm_wmma_kernel(const half_t* __restrict__ A,
                      const half_t* __restrict__ W,
                      const float* __restrict__ bias,
                      float* __restrict__ Y,
                      int M, int K, int Cout) {
  extern __shared__ char smem_raw[];          // 16 * K halves (<= 48KB)
  half_t* ldsW = (half_t*)smem_raw;

  int tn   = blockIdx.y;
  int wave = threadIdx.x >> 5;
  int lane = threadIdx.x & 31;
  int l15  = lane & 15;
  int khi  = (lane >> 4) << 3;                // 0 (lanes 0-15) / 8 (lanes 16-31)
  int tm   = blockIdx.x * 4 + wave;

  // ---- async DMA: stage W rows [tn*16 .. tn*16+15] x K into LDS ----
  {
    const char* gsrc   = (const char*)(W + (size_t)(tn * 16) * K);
    unsigned    ldsoff = (unsigned)(size_t)ldsW;   // LDS byte address
    int nchunks = (K * 32) >> 4;                   // 16-byte chunks
    for (int ch = threadIdx.x; ch < nchunks; ch += 128) {
      unsigned    loff = ldsoff + (unsigned)(ch << 4);
      const char* ga   = gsrc + ((size_t)ch << 4);
      asm volatile("global_load_async_to_lds_b128 %0, %1, off"
                   :: "v"(loff), "v"(ga)
                   : "memory");
    }
    asm volatile("s_wait_asynccnt 0x0" ::: "memory");
  }
  __syncthreads();

  const half_t* arow = A + (size_t)(tm * 16 + l15) * K;
  const half_t* wrow = ldsW + (size_t)l15 * K;

  v8f acc = {};
  for (int k0 = 0; k0 < K; k0 += 32) {
    __builtin_prefetch(arow + k0 + 128, 0, 3);
    v8h alo = *(const v8h*)(arow + k0 + khi);        // K = khi .. khi+7
    v8h ahi = *(const v8h*)(arow + k0 + 16 + khi);   // K = 16+khi .. 16+khi+7
    v8h blo = *(const v8h*)(wrow + k0 + khi);
    v8h bhi = *(const v8h*)(wrow + k0 + 16 + khi);
    v16h av = __builtin_shufflevector(alo, ahi, 0,1,2,3,4,5,6,7,8,9,10,11,12,13,14,15);
    v16h bv = __builtin_shufflevector(blo, bhi, 0,1,2,3,4,5,6,7,8,9,10,11,12,13,14,15);
    acc = __builtin_amdgcn_wmma_f32_16x16x32_f16(
        /*neg_a=*/false, av, /*neg_b=*/false, bv,
        /*c_mod=*/(short)0, acc, /*reuse_a=*/false, /*reuse_b=*/false);
  }
  int col = tn * 16 + l15;
  float bvv = bias ? bias[col] : 0.f;
  for (int r = 0; r < 8; ++r) {
    int row = tm * 16 + r + khi;      // lanes 16-31 hold M = r+8
    Y[(size_t)row * Cout + col] = acc[r] + bvv;
  }
}

// ------------------------- instance-norm pieces ----------------------------
__global__ __launch_bounds__(256)
void channel_stats_kernel(const float* __restrict__ Y, float* __restrict__ mv,
                          int M, int C) {
  int c = blockIdx.x;
  __shared__ float ss[256], sq[256];
  float s = 0.f, q = 0.f;
  for (int i = threadIdx.x; i < M; i += 256) {
    float v = Y[(size_t)i * C + c];
    s += v; q += v * v;
  }
  ss[threadIdx.x] = s; sq[threadIdx.x] = q;
  __syncthreads();
  for (int o = 128; o > 0; o >>= 1) {
    if (threadIdx.x < o) {
      ss[threadIdx.x] += ss[threadIdx.x + o];
      sq[threadIdx.x] += sq[threadIdx.x + o];
    }
    __syncthreads();
  }
  if (threadIdx.x == 0) {
    float m = ss[0] / (float)M;
    mv[c] = m;
    mv[C + c] = sq[0] / (float)M - m * m;
  }
}

// (x-m)*rsqrt(v+eps)*nw+nb, relu; emit f16 (next WMMA input) and/or f32
__global__ void norm_relu_kernel(const float* __restrict__ Y,
                                 const float* __restrict__ mv,
                                 const float* __restrict__ nw,
                                 const float* __restrict__ nb,
                                 int M, int C,
                                 half_t* __restrict__ outh, int ldh,
                                 float* __restrict__ outf) {
  size_t i = (size_t)blockIdx.x * blockDim.x + threadIdx.x;
  if (i >= (size_t)M * C) return;
  int c = (int)(i % C);
  size_t r = i / C;
  float v = (Y[i] - mv[c]) * rsqrtf(mv[C + c] + EPSN);
  if (nw) v = v * nw[c] + nb[c];
  v = fmaxf(v, 0.f);
  if (outh) outh[r * (size_t)ldh + c] = (half_t)v;
  if (outf) outf[i] = v;
}

__global__ void maxpool32_kernel(const float* __restrict__ feat,
                                 float* __restrict__ out, int s, int C) {
  int idx = blockIdx.x * blockDim.x + threadIdx.x;
  if (idx >= s * C) return;
  int g = idx / C, c = idx % C;
  float m = -3.4e38f;
  for (int k = 0; k < 32; ++k)
    m = fmaxf(m, feat[((size_t)g * 32 + k) * C + c]);
  out[idx] = m;
}

// --------------------- FP: 3-NN interpolate + concat -----------------------
__global__ void fp_interp_kernel(const float* __restrict__ xyzn,
                                 const float* __restrict__ xyzs,
                                 const float* __restrict__ fsrc,  // s x C2
                                 const float* __restrict__ fskip, // n x C1 (opt)
                                 int n, int s, int C1, int C2, int Kp,
                                 half_t* __restrict__ A) {
  int i = blockIdx.x;
  __shared__ int   s_ind[3];
  __shared__ float s_w[3];
  if (threadIdx.x == 0) {
    float px = xyzn[i * 3 + 0], py = xyzn[i * 3 + 1], pz = xyzn[i * 3 + 2];
    float d0 = 1e30f, d1 = 1e30f, d2 = 1e30f;
    int i0 = 0, i1 = 0, i2 = 0;
    for (int j = 0; j < s; ++j) {
      float dx = xyzs[j * 3 + 0] - px;
      float dy = xyzs[j * 3 + 1] - py;
      float dz = xyzs[j * 3 + 2] - pz;
      float d = dx * dx + dy * dy + dz * dz;
      if (d < d0)      { d2 = d1; i2 = i1; d1 = d0; i1 = i0; d0 = d; i0 = j; }
      else if (d < d1) { d2 = d1; i2 = i1; d1 = d;  i1 = j; }
      else if (d < d2) { d2 = d;  i2 = j; }
    }
    float r0 = 1.f / (d0 + 1e-8f);
    float r1 = 1.f / (d1 + 1e-8f);
    float r2 = 1.f / (d2 + 1e-8f);
    float rs = r0 + r1 + r2;
    s_ind[0] = i0; s_ind[1] = i1; s_ind[2] = i2;
    s_w[0] = r0 / rs; s_w[1] = r1 / rs; s_w[2] = r2 / rs;
  }
  __syncthreads();
  half_t* a = A + (size_t)i * Kp;
  for (int c = threadIdx.x; c < C1; c += blockDim.x)
    a[c] = (half_t)fskip[(size_t)i * C1 + c];
  for (int c = threadIdx.x; c < C2; c += blockDim.x) {
    float v = s_w[0] * fsrc[(size_t)s_ind[0] * C2 + c] +
              s_w[1] * fsrc[(size_t)s_ind[1] * C2 + c] +
              s_w[2] * fsrc[(size_t)s_ind[2] * C2 + c];
    a[C1 + c] = (half_t)v;
  }
}

// ------------------------------- host side ---------------------------------

extern "C" void kernel_launch(void* const* d_in, const int* in_sizes, int n_in,
                              void* d_out, int out_size, void* d_ws, size_t ws_size,
                              hipStream_t stream) {
  const float* verts = (const float*)d_in[0];
  const float* norms = (const float*)d_in[1];

  // Param leaves in JAX pytree (sorted-key) order after vertices/normals:
  // conv1{b,w}=2,3 ; fp1 L0..L2 ; fp2 L0..L1 ; fp3 ; fp4 ; sa1 L0..L2 ; sa2 ; sa3 ; sa4
  // each MLP layer contributes {b, nb, nw, w} (sorted).
  auto P = [&](int i) { return (const float*)d_in[i]; };
  const int CONV_B = 2, CONV_W = 3;
  const int FP1 = 4, FP2 = 16, FP3 = 24, FP4 = 32;
  const int SA1 = 40, SA2 = 52, SA3 = 64, SA4 = 76;

  // workspace bump allocator
  char* ws = (char*)d_ws;
  size_t off = 0;
  auto alloc = [&](size_t bytes) -> void* {
    void* p = ws + off;
    off = (off + bytes + 255) & ~(size_t)255;
    return p;
  };

  float* p0   = (float*)alloc(8192 * 6 * 4);
  float* xyz1 = (float*)alloc(2048 * 3 * 4);
  float* xyz2 = (float*)alloc(512 * 3 * 4);
  float* xyz3 = (float*)alloc(128 * 3 * 4);
  float* xyz4 = (float*)alloc(32 * 3 * 4);
  int*   idx1 = (int*)alloc(2048 * 4);
  int*   idx2 = (int*)alloc(512 * 4);
  int*   idx3 = (int*)alloc(128 * 4);
  int*   idx4 = (int*)alloc(32 * 4);
  int*   gidx = (int*)alloc(65536 * 4);
  float* f1   = (float*)alloc((size_t)2048 * 128 * 4);
  float* f2   = (float*)alloc((size_t)512 * 256 * 4);
  float* f3   = (float*)alloc((size_t)128 * 512 * 4);
  float* f4   = (float*)alloc((size_t)32 * 1024 * 4);
  float* f3n  = (float*)alloc((size_t)128 * 512 * 4);
  float* f2n  = (float*)alloc((size_t)512 * 256 * 4);
  float* f1n  = (float*)alloc((size_t)2048 * 128 * 4);
  half_t* A0  = (half_t*)alloc((size_t)4194304 * 2);   // 65536 x 64 max
  half_t* A1  = (half_t*)alloc((size_t)4194304 * 2);
  half_t* Wh  = (half_t*)alloc((size_t)1310720 * 2);   // 768 x 1536 max
  float* Ybuf = (float*)alloc((size_t)8388608 * 4);    // 65536 x 128 max
  float* mv   = (float*)alloc(2048 * 4);               // mean | var

  // one MLP layer: convert W, WMMA GEMM (+bias), per-channel stats, norm+relu
  auto mlp_layer = [&](const half_t* A, int M, int K, int lbase, int Cin, int Cout,
                       half_t* outh, int ldh, float* outf) {
    int tot = Cout * K;
    convert_w_kernel<<<(tot + 255) / 256, 256, 0, stream>>>(P(lbase + 3), Wh, Cout, Cin, K);
    dim3 g(M / 64, Cout / 16);
    size_t shmem = (size_t)K * 16 * sizeof(half_t);   // weight panel in LDS
    gemm_wmma_kernel<<<g, 128, shmem, stream>>>(A, Wh, P(lbase + 0), Ybuf, M, K, Cout);
    channel_stats_kernel<<<Cout, 256, 0, stream>>>(Ybuf, mv, M, Cout);
    size_t n = (size_t)M * Cout;
    norm_relu_kernel<<<(unsigned)((n + 255) / 256), 256, 0, stream>>>(
        Ybuf, mv, P(lbase + 2), P(lbase + 1), M, Cout, outh, ldh, outf);
  };

  // one set-abstraction stage
  auto sa_level = [&](const float* xyz, const float* pts, int n, int C, float r2,
                      float* new_xyz, int* fidx, int s, int lbase, int K0,
                      int c0, int c1, int c2, float* fout) {
    fps_kernel<<<1, 1024, 0, stream>>>(xyz, n, s, fidx, new_xyz);
    ball_query_kernel<<<(s + 63) / 64, 64, 0, stream>>>(xyz, new_xyz, n, s, r2, gidx);
    int M = s * 32;
    zero_h_kernel<<<1024, 256, 0, stream>>>(A0, (size_t)M * K0);
    group_gather_kernel<<<M, 64, 0, stream>>>(xyz, new_xyz, pts, gidx, s, C, K0, A0);
    mlp_layer(A0, M, K0, lbase,     C + 3, c0, A1, c0, nullptr);
    mlp_layer(A1, M, c0, lbase + 4, c0,    c1, A0, c1, nullptr);
    mlp_layer(A0, M, c1, lbase + 8, c1,    c2, nullptr, 0, Ybuf);  // in-place f32
    maxpool32_kernel<<<(s * c2 + 255) / 256, 256, 0, stream>>>(Ybuf, fout, s, c2);
  };

  // one 2-layer feature-propagation stage
  auto fp_level = [&](const float* xyzn, const float* xyzs,
                      const float* fskip, int C1, const float* fsrc, int C2,
                      int n, int s, int lbase, int c0, int c1, float* fout) {
    int K0 = C1 + C2;
    fp_interp_kernel<<<n, 64, 0, stream>>>(xyzn, xyzs, fsrc, fskip, n, s, C1, C2, K0, A0);
    mlp_layer(A0, n, K0, lbase,     K0, c0, A1, c0, nullptr);
    mlp_layer(A1, n, c0, lbase + 4, c0, c1, nullptr, 0, fout);
  };

  // -------- forward pass --------
  concat_p0_kernel<<<(8192 * 6 + 255) / 256, 256, 0, stream>>>(verts, norms, p0, 8192);

  sa_level(verts, p0, 8192,   6, 0.06f * 0.06f, xyz1, idx1, 2048, SA1,  32,  64,  64,  128, f1);
  sa_level(xyz1,  f1, 2048, 128, 0.10f * 0.10f, xyz2, idx2,  512, SA2, 160, 128, 128,  256, f2);
  sa_level(xyz2,  f2,  512, 256, 0.14f * 0.14f, xyz3, idx3,  128, SA3, 288, 256, 256,  512, f3);
  sa_level(xyz3,  f3,  128, 512, 0.18f * 0.18f, xyz4, idx4,   32, SA4, 544, 512, 512, 1024, f4);

  fp_level(xyz3, xyz4, f3, 512, f4, 1024,  128,  32, FP4, 768, 512, f3n);
  fp_level(xyz2, xyz3, f2, 256, f3n, 512,  512, 128, FP3, 512, 256, f2n);
  fp_level(xyz1, xyz2, f1, 128, f2n, 256, 2048, 512, FP2, 256, 128, f1n);

  // FP1: interp only (no skip), 3 layers 128->128->128->64
  fp_interp_kernel<<<8192, 64, 0, stream>>>(verts, xyz1, f1n, nullptr,
                                            8192, 2048, 0, 128, 128, A0);
  mlp_layer(A0, 8192, 128, FP1,     128, 128, A1,  128, nullptr);
  mlp_layer(A1, 8192, 128, FP1 + 4, 128, 128, A0,  128, nullptr);
  mlp_layer(A0, 8192, 128, FP1 + 8, 128,  64, A1,   64, nullptr);

  // conv1: f0 @ W^T + b, then instance-norm (no affine) + relu -> d_out
  convert_w_kernel<<<(64 * 64 + 255) / 256, 256, 0, stream>>>(P(CONV_W), Wh, 64, 64, 64);
  {
    dim3 g(8192 / 64, 64 / 16);
    gemm_wmma_kernel<<<g, 128, (size_t)64 * 16 * sizeof(half_t), stream>>>(
        A1, Wh, P(CONV_B), Ybuf, 8192, 64, 64);
  }
  channel_stats_kernel<<<64, 256, 0, stream>>>(Ybuf, mv, 8192, 64);
  {
    size_t n = (size_t)8192 * 64;
    norm_relu_kernel<<<(unsigned)((n + 255) / 256), 256, 0, stream>>>(
        Ybuf, mv, nullptr, nullptr, 8192, 64, nullptr, 0, (float*)d_out);
  }
}
